// MoELayer_78606491452341
// MI455X (gfx1250) — matrix operand order repaired and checked by reading, not compiled, not run on hardware
//
#include <hip/hip_runtime.h>
#include <hip/hip_bf16.h>

#define HIDDEN 1024
#define INTER  2816
#define NEXP   8
#define NTOK   8192
#define NROWS  (NTOK * 2)

typedef __attribute__((ext_vector_type(16))) __bf16 v16bf;
typedef __attribute__((ext_vector_type(8)))  __bf16 v8bf;
typedef __attribute__((ext_vector_type(8)))  float  v8f;

static __device__ __forceinline__ v16bf cat16(v8bf lo, v8bf hi) {
    return __builtin_shufflevector(lo, hi, 0,1,2,3,4,5,6,7,8,9,10,11,12,13,14,15);
}

// ---- gfx1250 async global->LDS path (guarded; falls back to VGPR staging) ----
#if defined(__has_builtin)
#if __has_builtin(__builtin_amdgcn_global_load_async_to_lds_b128) && \
    __has_builtin(__builtin_amdgcn_s_wait_asynccnt)
#define USE_ASYNC_LDS 1
#endif
#endif
#ifndef USE_ASYNC_LDS
#define USE_ASYNC_LDS 0
#endif

#if USE_ASYNC_LDS
typedef int v4i_ __attribute__((vector_size(16)));
typedef __attribute__((address_space(1))) v4i_ gv4i;
typedef __attribute__((address_space(3))) v4i_ lv4i;

static __device__ __forceinline__ void async_cp16(const __bf16* g, __bf16* l) {
    __builtin_amdgcn_global_load_async_to_lds_b128((gv4i*)g, (lv4i*)l, 0, 0);
}
#endif

// ---------------- utility kernels ----------------

__global__ void zero_f32_kernel(float* p, int n) {
    for (int i = blockIdx.x * blockDim.x + threadIdx.x; i < n; i += gridDim.x * blockDim.x)
        p[i] = 0.f;
}

__global__ void zero_i32_kernel(int* p, int n) {
    int i = blockIdx.x * blockDim.x + threadIdx.x;
    if (i < n) p[i] = 0;
}

__global__ void cvt_cast_kernel(const float* __restrict__ in, __bf16* __restrict__ out, int n) {
    for (int i = blockIdx.x * blockDim.x + threadIdx.x; i < n; i += gridDim.x * blockDim.x)
        out[i] = (__bf16)in[i];
}

// in:  [NEXP][R][C] f32   ->   out: [NEXP][C][R] bf16  (K-major for GEMM B tiles)
__global__ void cvt_transpose_kernel(const float* __restrict__ in, __bf16* __restrict__ out,
                                     int R, int C) {
    long total = (long)NEXP * R * C;
    long i = (long)blockIdx.x * blockDim.x + threadIdx.x;
    if (i >= total) return;
    long rc = (long)R * C;
    int e = (int)(i / rc);
    long rem = i - (long)e * rc;
    int c = (int)(rem / R);
    int r = (int)(rem - (long)c * R);
    out[i] = (__bf16)in[(long)e * rc + (long)r * C + c];
}

// ---------------- router: one wave32 per token ----------------

__global__ __launch_bounds__(256) void router_kernel(const float* __restrict__ x,
                                                     const float* __restrict__ gw,
                                                     int* __restrict__ tokE,
                                                     float* __restrict__ tokW) {
    int lane = threadIdx.x & 31;
    int wave = threadIdx.x >> 5;
    int tok  = blockIdx.x * 8 + wave;
    if (tok >= NTOK) return;

    const float* xr = x + (size_t)tok * HIDDEN;
    float part[NEXP];
#pragma unroll
    for (int e = 0; e < NEXP; ++e) part[e] = 0.f;

    for (int d = lane; d < HIDDEN; d += 32) {
        float xv = xr[d];
#pragma unroll
        for (int e = 0; e < NEXP; ++e) part[e] += xv * gw[e * HIDDEN + d];
    }
#pragma unroll
    for (int e = 0; e < NEXP; ++e) {
        float v = part[e];
#pragma unroll
        for (int m = 16; m > 0; m >>= 1) v += __shfl_xor(v, m, 32);
        part[e] = v;
    }
    if (lane == 0) {
        int e0 = 0; float b0 = part[0];
#pragma unroll
        for (int e = 1; e < NEXP; ++e) if (part[e] > b0) { b0 = part[e]; e0 = e; }
        int e1 = -1; float b1 = -1e30f;
#pragma unroll
        for (int e = 0; e < NEXP; ++e) if (e != e0 && part[e] > b1) { b1 = part[e]; e1 = e; }
        float p0 = 1.f / (1.f + __expf(b1 - b0));   // softmax over {b0,b1}, b0>=b1
        tokE[2 * tok]     = e0;
        tokE[2 * tok + 1] = e1;
        tokW[2 * tok]     = p0;
        tokW[2 * tok + 1] = 1.f - p0;
    }
}

__global__ void count_kernel(const int* __restrict__ tokE, int* __restrict__ counts) {
    int i = blockIdx.x * blockDim.x + threadIdx.x;
    if (i < NROWS) atomicAdd(&counts[tokE[i]], 1);
}

__global__ void scan_kernel(const int* __restrict__ counts, int* __restrict__ offs) {
    if (threadIdx.x == 0 && blockIdx.x == 0) {
        int s = 0;
        for (int e = 0; e < NEXP; ++e) { offs[e] = s; s += counts[e]; }
        offs[NEXP] = s;
    }
}

__global__ void scatter_kernel(const int* __restrict__ tokE, const float* __restrict__ tokW,
                               const int* __restrict__ offs, int* __restrict__ cursors,
                               int* __restrict__ rowTok, float* __restrict__ rowW) {
    int i = blockIdx.x * blockDim.x + threadIdx.x;
    if (i >= NROWS) return;
    int e = tokE[i];
    int slot = atomicAdd(&cursors[e], 1);
    int dst = offs[e] + slot;
    rowTok[dst] = i >> 1;
    rowW[dst] = tokW[i];
}

// ---------------- shared WMMA mainloop ----------------
// Block: 256 thr (8 waves).  Tile: 128x128, K-chunk 32, double-buffered LDS.
// Wave w: wm = w&3 (32-row strip), wn = w>>2 (64-col strip); 8 accumulators.
// Per thread staging: row/col = t>>1, kg = (t&1)*16 (two b128 per array per chunk).

template <int KDIM>
static __device__ __forceinline__ void mma_mainloop(
    const __bf16* __restrict__ aSrc,   // thread base: row ptr + kg
    const __bf16* __restrict__ bSrc,   // thread base: col ptr + kg
    __bf16 (*Abuf)[128][32], __bf16 (*Bbuf)[128][32],
    int aRowT, int kg, int wm, int wn, int lane, v8f acc[2][4]) {

    constexpr int NCH = KDIM / 32;
    const int kbA = (lane >> 4) * 8;
    const int kbB = (lane >> 4) * 8;

#if !USE_ASYNC_LDS
    v8bf pa0, pa1, pb0, pb1;
#endif

    // stage chunk 0 into buffer 0
#if USE_ASYNC_LDS
    async_cp16(aSrc,     &Abuf[0][aRowT][kg]);
    async_cp16(aSrc + 8, &Abuf[0][aRowT][kg + 8]);
    async_cp16(bSrc,     &Bbuf[0][aRowT][kg]);
    async_cp16(bSrc + 8, &Bbuf[0][aRowT][kg + 8]);
    __builtin_amdgcn_s_wait_asynccnt(0);
#else
    pa0 = *(const v8bf*)(aSrc);
    pa1 = *(const v8bf*)(aSrc + 8);
    pb0 = *(const v8bf*)(bSrc);
    pb1 = *(const v8bf*)(bSrc + 8);
    *(v8bf*)&Abuf[0][aRowT][kg]     = pa0;
    *(v8bf*)&Abuf[0][aRowT][kg + 8] = pa1;
    *(v8bf*)&Bbuf[0][aRowT][kg]     = pb0;
    *(v8bf*)&Bbuf[0][aRowT][kg + 8] = pb1;
#endif
    __syncthreads();

    for (int ch = 0; ch < NCH; ++ch) {
        const int cur = ch & 1;
        const int nxt = cur ^ 1;
        const bool more = (ch + 1) < NCH;
        const int k0n = (ch + 1) * 32;

        // kick off next chunk while we compute on the current one
        if (more) {
#if USE_ASYNC_LDS
            async_cp16(aSrc + k0n,     &Abuf[nxt][aRowT][kg]);
            async_cp16(aSrc + k0n + 8, &Abuf[nxt][aRowT][kg + 8]);
            async_cp16(bSrc + k0n,     &Bbuf[nxt][aRowT][kg]);
            async_cp16(bSrc + k0n + 8, &Bbuf[nxt][aRowT][kg + 8]);
#else
            pa0 = *(const v8bf*)(aSrc + k0n);
            pa1 = *(const v8bf*)(aSrc + k0n + 8);
            pb0 = *(const v8bf*)(bSrc + k0n);
            pb1 = *(const v8bf*)(bSrc + k0n + 8);
#endif
        }

        // 8 WMMAs from buffer `cur`
        v16bf a[2];
#pragma unroll
        for (int mi = 0; mi < 2; ++mi) {
            int rowA = wm * 32 + mi * 16 + (lane & 15);
            a[mi] = cat16(*(const v8bf*)&Abuf[cur][rowA][kbA],
                          *(const v8bf*)&Abuf[cur][rowA][16 + kbA]);
        }
#pragma unroll
        for (int ni = 0; ni < 4; ++ni) {
            int colB = wn * 64 + ni * 16 + (lane & 15);
            v16bf b = cat16(*(const v8bf*)&Bbuf[cur][colB][kbB],
                            *(const v8bf*)&Bbuf[cur][colB][16 + kbB]);
#pragma unroll
            for (int mi = 0; mi < 2; ++mi)
                acc[mi][ni] = __builtin_amdgcn_wmma_f32_16x16x32_bf16(
                    false, a[mi], false, b, (short)0, acc[mi][ni], false, false);
        }

        if (more) {
#if USE_ASYNC_LDS
            __builtin_amdgcn_s_wait_asynccnt(0);
#else
            *(v8bf*)&Abuf[nxt][aRowT][kg]     = pa0;
            *(v8bf*)&Abuf[nxt][aRowT][kg + 8] = pa1;
            *(v8bf*)&Bbuf[nxt][aRowT][kg]     = pb0;
            *(v8bf*)&Bbuf[nxt][aRowT][kg + 8] = pb1;
#endif
        }
        __syncthreads();   // buf[nxt] ready; everyone done reading buf[cur]
    }
}

// ---------------- grouped GEMM1 + SiLU : act = silu(H_e @ W1_e) ----------------

__global__ __launch_bounds__(256) void gemm1_silu_kernel(
    const __bf16* __restrict__ xbf, const __bf16* __restrict__ w1T,
    const int* __restrict__ rowTok, const int* __restrict__ counts,
    const int* __restrict__ offs, __bf16* __restrict__ act) {

    int e  = blockIdx.z;
    int ne = counts[e];
    int m0 = blockIdx.y * 128;
    if (m0 >= ne) return;
    int n0  = blockIdx.x * 128;
    int off = offs[e];

    __shared__ __align__(16) __bf16 Abuf[2][128][32];
    __shared__ __align__(16) __bf16 Bbuf[2][128][32];

    int t     = threadIdx.x;
    int lane  = t & 31;
    int wave  = t >> 5;
    int aRowT = t >> 1;           // 0..127 : A row / B col staged by this thread
    int kg    = (t & 1) * 16;     // K sub-offset

    int r   = m0 + aRowT;
    int tok = rowTok[off + (r < ne ? r : 0)];
    const __bf16* aSrc = xbf + (size_t)tok * HIDDEN + kg;
    const __bf16* bSrc = w1T + ((size_t)e * INTER + (n0 + aRowT)) * HIDDEN + kg;

    int wm = wave & 3;
    int wn = wave >> 2;

    v8f acc[2][4];
#pragma unroll
    for (int mi = 0; mi < 2; ++mi)
#pragma unroll
        for (int ni = 0; ni < 4; ++ni)
            acc[mi][ni] = (v8f){0.f,0.f,0.f,0.f,0.f,0.f,0.f,0.f};

    mma_mainloop<HIDDEN>(aSrc, bSrc, Abuf, Bbuf, aRowT, kg, wm, wn, lane, acc);

    // C/D layout: VGPR v -> M = v (lanes 0-15) or v+8 (lanes 16-31); N = lane&15
    int rhalf = (lane >> 4) << 3;
#pragma unroll
    for (int mi = 0; mi < 2; ++mi) {
#pragma unroll
        for (int v = 0; v < 8; ++v) {
            int rr = m0 + wm * 32 + mi * 16 + rhalf + v;
            if (rr < ne) {
                size_t rp = (size_t)(off + rr) * INTER;
#pragma unroll
                for (int ni = 0; ni < 4; ++ni) {
                    int c = n0 + wn * 64 + ni * 16 + (lane & 15);
                    float a0 = acc[mi][ni][v];
                    act[rp + c] = (__bf16)(a0 / (1.f + __expf(-a0)));
                }
            }
        }
    }
}

// ---------------- grouped GEMM2 + scale + scatter-add ----------------

__global__ __launch_bounds__(256) void gemm2_scatter_kernel(
    const __bf16* __restrict__ act, const __bf16* __restrict__ w2T,
    const int* __restrict__ rowTok, const float* __restrict__ rowW,
    const int* __restrict__ counts, const int* __restrict__ offs,
    float* __restrict__ out) {

    int e  = blockIdx.z;
    int ne = counts[e];
    int m0 = blockIdx.y * 128;
    if (m0 >= ne) return;
    int n0  = blockIdx.x * 128;
    int off = offs[e];

    __shared__ __align__(16) __bf16 Abuf[2][128][32];
    __shared__ __align__(16) __bf16 Bbuf[2][128][32];

    int t     = threadIdx.x;
    int lane  = t & 31;
    int wave  = t >> 5;
    int aRowT = t >> 1;
    int kg    = (t & 1) * 16;

    int r = m0 + aRowT;
    const __bf16* aSrc = act + (size_t)(off + (r < ne ? r : 0)) * INTER + kg;
    const __bf16* bSrc = w2T + ((size_t)e * HIDDEN + (n0 + aRowT)) * INTER + kg;

    int wm = wave & 3;
    int wn = wave >> 2;

    v8f acc[2][4];
#pragma unroll
    for (int mi = 0; mi < 2; ++mi)
#pragma unroll
        for (int ni = 0; ni < 4; ++ni)
            acc[mi][ni] = (v8f){0.f,0.f,0.f,0.f,0.f,0.f,0.f,0.f};

    mma_mainloop<INTER>(aSrc, bSrc, Abuf, Bbuf, aRowT, kg, wm, wn, lane, acc);

    int rhalf = (lane >> 4) << 3;
#pragma unroll
    for (int mi = 0; mi < 2; ++mi) {
#pragma unroll
        for (int v = 0; v < 8; ++v) {
            int rr = m0 + wm * 32 + mi * 16 + rhalf + v;
            if (rr < ne) {
                int gi = off + rr;
                float wsc = rowW[gi];
                float* yp = out + (size_t)rowTok[gi] * HIDDEN;
#pragma unroll
                for (int ni = 0; ni < 4; ++ni) {
                    int c = n0 + wn * 64 + ni * 16 + (lane & 15);
                    __hip_atomic_fetch_add(&yp[c], wsc * acc[mi][ni][v],
                                           __ATOMIC_RELAXED, __HIP_MEMORY_SCOPE_AGENT);
                }
            }
        }
    }
}

// ---------------- launch ----------------

extern "C" void kernel_launch(void* const* d_in, const int* in_sizes, int n_in,
                              void* d_out, int out_size, void* d_ws, size_t ws_size,
                              hipStream_t stream) {
    const float* x  = (const float*)d_in[0];
    const float* gw = (const float*)d_in[1];
    const float* w1 = (const float*)d_in[2];
    const float* w2 = (const float*)d_in[3];
    float* out = (float*)d_out;

    char* ws = (char*)d_ws;
    size_t o = 0;
    auto alloc = [&](size_t bytes) -> char* {
        char* p = ws + o;
        o += (bytes + 255) & ~(size_t)255;
        return p;
    };
    __bf16* xbf    = (__bf16*)alloc((size_t)NTOK * HIDDEN * 2);
    __bf16* w1T    = (__bf16*)alloc((size_t)NEXP * INTER * HIDDEN * 2);
    __bf16* w2T    = (__bf16*)alloc((size_t)NEXP * HIDDEN * INTER * 2);
    __bf16* act    = (__bf16*)alloc((size_t)NROWS * INTER * 2);
    int*    tokE   = (int*)alloc(NROWS * 4);
    float*  tokW   = (float*)alloc(NROWS * 4);
    int*    rowTok = (int*)alloc(NROWS * 4);
    float*  rowW   = (float*)alloc(NROWS * 4);
    int*    counts = (int*)alloc(64);
    int*    cursors= (int*)alloc(64);
    int*    offs   = (int*)alloc(64);

    zero_f32_kernel<<<2048, 256, 0, stream>>>(out, NTOK * HIDDEN);
    zero_i32_kernel<<<1, 64, 0, stream>>>(counts, NEXP);
    zero_i32_kernel<<<1, 64, 0, stream>>>(cursors, NEXP);

    cvt_cast_kernel<<<4096, 256, 0, stream>>>(x, xbf, NTOK * HIDDEN);
    {
        long tot = (long)NEXP * HIDDEN * INTER;
        int blocks = (int)((tot + 255) / 256);
        cvt_transpose_kernel<<<blocks, 256, 0, stream>>>(w1, w1T, HIDDEN, INTER);
        cvt_transpose_kernel<<<blocks, 256, 0, stream>>>(w2, w2T, INTER, HIDDEN);
    }

    router_kernel<<<NTOK / 8, 256, 0, stream>>>(x, gw, tokE, tokW);
    count_kernel<<<NROWS / 256, 256, 0, stream>>>(tokE, counts);
    scan_kernel<<<1, 1, 0, stream>>>(counts, offs);
    scatter_kernel<<<NROWS / 256, 256, 0, stream>>>(tokE, tokW, offs, cursors, rowTok, rowW);

    gemm1_silu_kernel<<<dim3(INTER / 128, 64, NEXP), 256, 0, stream>>>(
        xbf, w1T, rowTok, counts, offs, act);
    gemm2_scatter_kernel<<<dim3(HIDDEN / 128, 64, NEXP), 256, 0, stream>>>(
        act, w2T, rowTok, rowW, counts, offs, out);

    (void)in_sizes; (void)n_in; (void)out_size; (void)ws_size;
}
